// DescriptorNetworkTorch_51058571215438
// MI455X (gfx1250) — matrix operand order repaired and checked by reading, not compiled, not run on hardware
//
#include <hip/hip_runtime.h>
#include <hip/hip_bf16.h>

typedef _Float16 half_t;
typedef __attribute__((ext_vector_type(16))) _Float16 v16h;
typedef __attribute__((ext_vector_type(8)))  _Float16 v8h;
typedef __attribute__((ext_vector_type(8)))  float    v8f;

#define GG   256
#define KK   16
#define LL   3
#define HH   3
#define FF   64
#define EMBD 200
#define HIDD 256
#define NN   (GG*KK)           // 4096 nodes
#define EE   (NN*(KK-1))       // 61440 edges
#define EPN  (KK-1)            // 15 edges per node (contiguous by construction)

__device__ __forceinline__ float selu_f(float x) {
  const float kScale = 1.0507009873554805f;
  const float kAlpha = 1.6732632423543772f;
  return kScale * (x > 0.0f ? x : kAlpha * (__expf(x) - 1.0f));
}

// ---------------------------------------------------------------- utilities
__global__ void copy_f32_kernel(const float* __restrict__ src, float* __restrict__ dst, int n) {
  int i = blockIdx.x * blockDim.x + threadIdx.x;
  if (i < n) dst[i] = src[i];
}

__global__ void zero_f32_kernel(float* __restrict__ dst, int n) {
  int i = blockIdx.x * blockDim.x + threadIdx.x;
  if (i < n) dst[i] = 0.0f;
}

__global__ void cvt_f16_kernel(const float* __restrict__ src, half_t* __restrict__ dst, int n) {
  int i = blockIdx.x * blockDim.x + threadIdx.x;
  if (i < n) dst[i] = (half_t)src[i];
}

// ------------------------------------------------- weight pack (f32 -> f16)
// Packs B[K,N] (row-major, f32) into WMMA B-fragment-major f16 layout:
//   frag index = nt*ktiles + kt ; within fragment: lane 0..31 x 16 halves.
//   element (lane, i) = B[kt*32 + (lane&16) + i][nt*16 + (lane&15)]
// so each lane's fragment is one contiguous, 32B-aligned v16h load.
__global__ void pack_b_kernel(const float* __restrict__ src, half_t* __restrict__ dst,
                              int K, int N, int nmat) {
  int i = blockIdx.x * blockDim.x + threadIdx.x;
  int per = K * N;
  if (i >= nmat * per) return;
  int mat = i / per;
  int r   = i % per;
  int ktiles = K / 32;
  int frag = r >> 9;          // /512
  int q    = r & 511;
  int lane = q >> 4;
  int ii   = q & 15;
  int nt = frag / ktiles;
  int kt = frag % ktiles;
  int k = kt * 32 + (lane & 16) + ii;
  int n = nt * 16 + (lane & 15);
  dst[(size_t)mat * per + r] = (half_t)src[(size_t)mat * per + (size_t)k * N + n];
}

// ------------------------------------------------------------ initial embed
// fea[n,0:63] = x[n,:] @ W_init + b_init ; fea[n,63] = w[n]
__global__ void embed_kernel(const float* __restrict__ x, const float* __restrict__ W,
                             const float* __restrict__ b, const float* __restrict__ w,
                             float* __restrict__ fea) {
  int n = blockIdx.x;
  int f = threadIdx.x;                 // 64 threads
  if (f == FF - 1) { fea[n * FF + f] = w[n]; return; }
  const float* xr = x + (size_t)n * EMBD;
  float acc = b[f];
#pragma unroll 4
  for (int k = 0; k < EMBD; ++k) acc += xr[k] * W[k * (FF - 1) + f];
  fea[n * FF + f] = acc;
}

// -------------------------------------------------------------- pair build
// pair[e,0:64]=fea[self[e]] ; pair[e,64:128]=fea[nbr[e]]   (f32 -> f16)
__global__ void build_pair_kernel(const float* __restrict__ fea,
                                  const int* __restrict__ self_idx,
                                  const int* __restrict__ nbr_idx,
                                  half_t* __restrict__ pair) {
  int i = blockIdx.x * blockDim.x + threadIdx.x;
  if (i >= EE * 2 * FF) return;
  int e = i >> 7;                      // / 128
  int c = i & 127;
  int node = (c < FF) ? self_idx[e] : nbr_idx[e];
  pair[i] = (half_t)fea[node * FF + (c & (FF - 1))];
}

// ------------------------------------------------------------- WMMA GEMM
// C[M,N] = A[M,K] @ Bpack + bias[N], optional fused SELU with f16 output.
// A is row-major f16; Bpack is fragment-major (see pack_b_kernel).
// One wave per 16x64 strip; KS (= K/32) is compile-time so the whole
// load/WMMA schedule unrolls: 8 B-fragment b128 loads per k-step issue as a
// clause ahead of the 4-WMMA chain, letting the scheduler overlap k-steps.
// grid = (M/16, N/64), block = 32 (wave32).
union AFrag { v16h v; struct { v8h lo; v8h hi; } p; };

template <bool SELU_F16_OUT, int KS>
__global__ void gemm16_kernel(const half_t* __restrict__ A, int lda,
                              const half_t* __restrict__ Bpack,
                              const float* __restrict__ bias,
                              void* __restrict__ Cout, int ldc) {
  const int lane   = threadIdx.x & 31;
  const int lane15 = lane & 15;
  const int laneHi = lane >> 4;
  const int m0     = blockIdx.x * 16;
  const int ntile0 = blockIdx.y * 4;   // first of 4 n-tiles (64 columns)

  v8f c[4] = {};

  // A fragment: lanes 0-15 row=m0+lane hold K 0-15,
  // lanes 16-31 same rows hold K 16-31 (two 16B chunks per lane).
  const half_t* arow = A + (size_t)(m0 + lane15) * lda + laneHi * 16;
  const half_t* bbase = Bpack + ((size_t)ntile0 * KS * 32 + lane) * 16;

#pragma unroll
  for (int ks = 0; ks < KS; ++ks) {
    AFrag a;
    a.p.lo = *(const v8h*)(arow + ks * 32);
    a.p.hi = *(const v8h*)(arow + ks * 32 + 8);
    v16h b[4];
#pragma unroll
    for (int j = 0; j < 4; ++j)
      b[j] = *(const v16h*)(bbase + ((size_t)(j * KS + ks) * 32) * 16);
#pragma unroll
    for (int j = 0; j < 4; ++j)
      c[j] = __builtin_amdgcn_wmma_f32_16x16x32_f16(
          /*neg_a=*/false, a.v, /*neg_b=*/false, b[j],
          /*c_mod=*/(short)0, c[j], /*reuse_a=*/false, /*reuse_b=*/false);
  }

  // D layout: VGPR r, lanes 0-15 -> (M=r, N=lane); lanes 16-31 -> (M=8+r).
#pragma unroll
  for (int j = 0; j < 4; ++j) {
    const int col = (ntile0 + j) * 16 + lane15;
    const float bv = bias[col];
#pragma unroll
    for (int r = 0; r < 8; ++r) {
      const int row = m0 + r + laneHi * 8;
      float x = c[j][r] + bv;
      if (SELU_F16_OUT) {
        ((half_t*)Cout)[(size_t)row * ldc + col] = (half_t)selu_f(x);
      } else {
        ((float*)Cout)[(size_t)row * ldc + col] = x;
      }
    }
  }
}

// ------------------------------------------------------------- gate GEMV
// gate[r] = hid[r,:256] . W2 + b2   (one thread per row, K=256, v8h loads)
__global__ void gate_gemv_kernel(const half_t* __restrict__ hid,
                                 const float* __restrict__ W2,
                                 const float* __restrict__ b2,
                                 float* __restrict__ gate, int rows) {
  int r = blockIdx.x * blockDim.x + threadIdx.x;
  if (r >= rows) return;
  const v8h* h = (const v8h*)(hid + (size_t)r * HIDD);
  float acc = *b2;
#pragma unroll 4
  for (int kb = 0; kb < HIDD / 8; ++kb) {
    v8h hk = h[kb];
#pragma unroll
    for (int i = 0; i < 8; ++i) acc += (float)hk[i] * W2[kb * 8 + i];
  }
  gate[r] = acc;
}

// ----------------------------------------- per-node softmax + segment sum
// Edges for node n are contiguous: [n*15, n*15+15). 64 threads per node.
__global__ void attn_agg_msg_kernel(const float* __restrict__ gate,
                                    const float* __restrict__ msg,
                                    const float* __restrict__ w,
                                    const int* __restrict__ nbr_idx,
                                    const float* __restrict__ powp,
                                    float* __restrict__ fea_next, float invH) {
  int n = blockIdx.x;
  int f = threadIdx.x;                 // 64 threads
  const float pw = *powp;
  const int e0 = n * EPN;
  float m = -3.4e38f;
#pragma unroll
  for (int j = 0; j < EPN; ++j) m = fmaxf(m, gate[e0 + j]);
  float den = 1e-10f, acc = 0.0f;
#pragma unroll
  for (int j = 0; j < EPN; ++j) {
    float t = __powf(w[nbr_idx[e0 + j]], pw) * __expf(gate[e0 + j] - m);
    den += t;
    acc += t * msg[(size_t)(e0 + j) * FF + f];
  }
  fea_next[n * FF + f] += invH * acc / den;
}

// ---------------------------------------- per-graph softmax + segment sum
// Nodes for graph g are contiguous: [g*16, g*16+16). 64 threads per graph.
__global__ void attn_agg_cry_kernel(const float* __restrict__ gate,
                                    const float* __restrict__ msg,
                                    const float* __restrict__ w,
                                    const float* __restrict__ powp,
                                    float* __restrict__ out, float invH) {
  int g = blockIdx.x;
  int f = threadIdx.x;
  const float pw = *powp;
  const int n0 = g * KK;
  float m = -3.4e38f;
#pragma unroll
  for (int j = 0; j < KK; ++j) m = fmaxf(m, gate[n0 + j]);
  float den = 1e-10f, acc = 0.0f;
#pragma unroll
  for (int j = 0; j < KK; ++j) {
    float t = __powf(w[n0 + j], pw) * __expf(gate[n0 + j] - m);
    den += t;
    acc += t * msg[(size_t)(n0 + j) * FF + f];
  }
  out[g * FF + f] += invH * acc / den;
}

// ------------------------------------------------------------------ host
extern "C" void kernel_launch(void* const* d_in, const int* in_sizes, int n_in,
                              void* d_out, int out_size, void* d_ws, size_t ws_size,
                              hipStream_t stream) {
  (void)in_sizes; (void)n_in; (void)out_size; (void)ws_size;

  const float* elem_weights = (const float*)d_in[0];   // [N]
  const float* elem_fea_in  = (const float*)d_in[1];   // [N,200]
  const float* W_init       = (const float*)d_in[2];   // [200,63]
  const float* b_init       = (const float*)d_in[3];   // [63]
  const float* mg_W1        = (const float*)d_in[4];   // [L,H,128,256]
  const float* mg_b1        = (const float*)d_in[5];   // [L,H,256]
  const float* mg_W2        = (const float*)d_in[6];   // [L,H,256]
  const float* mg_b2        = (const float*)d_in[7];   // [L,H]
  const float* mm_W1        = (const float*)d_in[8];   // [L,H,128,256]
  const float* mm_b1        = (const float*)d_in[9];   // [L,H,256]
  const float* mm_W2        = (const float*)d_in[10];  // [L,H,256,64]
  const float* mm_b2        = (const float*)d_in[11];  // [L,H,64]
  const float* m_pow        = (const float*)d_in[12];  // [L,H]
  const float* cg_W1        = (const float*)d_in[13];  // [H,64,256]
  const float* cg_b1        = (const float*)d_in[14];  // [H,256]
  const float* cg_W2        = (const float*)d_in[15];  // [H,256]
  const float* cg_b2        = (const float*)d_in[16];  // [H]
  const float* cm_W1        = (const float*)d_in[17];  // [H,64,256]
  const float* cm_b1        = (const float*)d_in[18];  // [H,256]
  const float* cm_W2        = (const float*)d_in[19];  // [H,256,64]
  const float* cm_b2        = (const float*)d_in[20];  // [H,64]
  const float* c_pow        = (const float*)d_in[21];  // [H]
  const int*   self_idx     = (const int*)d_in[23];    // [E]
  const int*   nbr_idx      = (const int*)d_in[24];    // [E]

  // ---- workspace carve-up (all < L2; total ~67 MB)
  char* ws = (char*)d_ws;
  size_t off = 0;
  auto carve = [&](size_t bytes) -> char* {
    off = (off + 255) & ~(size_t)255;
    char* p = ws + off;
    off += bytes;
    return p;
  };
  half_t* mgW1h = (half_t*)carve((size_t)LL * HH * 2 * FF * HIDD * 2);
  half_t* mmW1h = (half_t*)carve((size_t)LL * HH * 2 * FF * HIDD * 2);
  half_t* mmW2h = (half_t*)carve((size_t)LL * HH * HIDD * FF * 2);
  half_t* cgW1h = (half_t*)carve((size_t)HH * FF * HIDD * 2);
  half_t* cmW1h = (half_t*)carve((size_t)HH * FF * HIDD * 2);
  half_t* cmW2h = (half_t*)carve((size_t)HH * HIDD * FF * 2);
  half_t* pairh = (half_t*)carve((size_t)EE * 2 * FF * 2);
  half_t* hidh  = (half_t*)carve((size_t)EE * HIDD * 2);
  float*  gateb = (float*)carve((size_t)EE * 4);
  float*  msgb  = (float*)carve((size_t)EE * FF * 4);
  float*  feaA  = (float*)carve((size_t)NN * FF * 4);
  float*  feaB  = (float*)carve((size_t)NN * FF * 4);
  half_t* feaH  = (half_t*)carve((size_t)NN * FF * 2);

  auto pack = [&](const float* s, half_t* d, int K, int N, int nmat) {
    int total = nmat * K * N;
    pack_b_kernel<<<(total + 255) / 256, 256, 0, stream>>>(s, d, K, N, nmat);
  };

  // ---- weight conversions + fragment packing (once per call)
  pack(mg_W1, mgW1h, 2 * FF, HIDD, LL * HH);
  pack(mm_W1, mmW1h, 2 * FF, HIDD, LL * HH);
  pack(mm_W2, mmW2h, HIDD, FF, LL * HH);
  pack(cg_W1, cgW1h, FF, HIDD, HH);
  pack(cm_W1, cmW1h, FF, HIDD, HH);
  pack(cm_W2, cmW2h, HIDD, FF, HH);

  // ---- initial embedding
  embed_kernel<<<NN, FF, 0, stream>>>(elem_fea_in, W_init, b_init, elem_weights, feaA);

  float* fea = feaA;
  float* fea_next = feaB;
  const float invH = 1.0f / HH;

  // ---- message-passing layers
  for (int l = 0; l < LL; ++l) {
    copy_f32_kernel<<<(NN * FF + 255) / 256, 256, 0, stream>>>(fea, fea_next, NN * FF);
    build_pair_kernel<<<(EE * 2 * FF + 255) / 256, 256, 0, stream>>>(fea, self_idx, nbr_idx, pairh);

    for (int h = 0; h < HH; ++h) {
      const int lh = l * HH + h;
      // gate hidden: selu(pair @ mg_W1 + b1) -> f16 [E,256]   (K=128 -> KS=4)
      gemm16_kernel<true, 4><<<dim3(EE / 16, HIDD / 64), 32, 0, stream>>>(
          pairh, 2 * FF, mgW1h + (size_t)lh * 2 * FF * HIDD,
          mg_b1 + (size_t)lh * HIDD, hidh, HIDD);
      // scalar gate: hid @ mg_W2 + b2 -> [E]
      gate_gemv_kernel<<<(EE + 255) / 256, 256, 0, stream>>>(
          hidh, mg_W2 + (size_t)lh * HIDD, mg_b2 + lh, gateb, EE);
      // message hidden: selu(pair @ mm_W1 + b1) -> f16 [E,256] (reuse hid)
      gemm16_kernel<true, 4><<<dim3(EE / 16, HIDD / 64), 32, 0, stream>>>(
          pairh, 2 * FF, mmW1h + (size_t)lh * 2 * FF * HIDD,
          mm_b1 + (size_t)lh * HIDD, hidh, HIDD);
      // message: hid @ mm_W2 + b2 -> f32 [E,64]               (K=256 -> KS=8)
      gemm16_kernel<false, 8><<<dim3(EE / 16, FF / 64), 32, 0, stream>>>(
          hidh, HIDD, mmW2h + (size_t)lh * HIDD * FF,
          mm_b2 + (size_t)lh * FF, msgb, FF);
      // softmax over 15 edges per node + weighted segment sum, residual add
      attn_agg_msg_kernel<<<NN, FF, 0, stream>>>(
          gateb, msgb, elem_weights, nbr_idx, m_pow + lh, fea_next, invH);
    }
    float* t = fea; fea = fea_next; fea_next = t;
  }

  // ---- crystal pooling (reuses the same kernels; N=4096 rows)
  cvt_f16_kernel<<<(NN * FF + 255) / 256, 256, 0, stream>>>(fea, feaH, NN * FF);
  zero_f32_kernel<<<(GG * FF + 255) / 256, 256, 0, stream>>>((float*)d_out, GG * FF);

  for (int h = 0; h < HH; ++h) {
    // K=64 -> KS=2
    gemm16_kernel<true, 2><<<dim3(NN / 16, HIDD / 64), 32, 0, stream>>>(
        feaH, FF, cgW1h + (size_t)h * FF * HIDD,
        cg_b1 + (size_t)h * HIDD, hidh, HIDD);
    gate_gemv_kernel<<<(NN + 255) / 256, 256, 0, stream>>>(
        hidh, cg_W2 + (size_t)h * HIDD, cg_b2 + h, gateb, NN);
    gemm16_kernel<true, 2><<<dim3(NN / 16, HIDD / 64), 32, 0, stream>>>(
        feaH, FF, cmW1h + (size_t)h * FF * HIDD,
        cm_b1 + (size_t)h * HIDD, hidh, HIDD);
    gemm16_kernel<false, 8><<<dim3(NN / 16, FF / 64), 32, 0, stream>>>(
        hidh, HIDD, cmW2h + (size_t)h * HIDD * FF,
        cm_b2 + (size_t)h * FF, msgb, FF);
    attn_agg_cry_kernel<<<GG, FF, 0, stream>>>(
        gateb, msgb, elem_weights, c_pow + h, (float*)d_out, invH);
  }
}
